// PerceiverAttention_6949257085329
// MI455X (gfx1250) — compile-verified, matrix operand order checked by reading, not run on hardware
//
#include <hip/hip_runtime.h>
#include <hip/hip_bf16.h>
#include <stdint.h>

// ---------------- problem constants ----------------
#define DIM       1024
#define DIM_HEAD  64
#define HEADS     8
#define INNER     512      // HEADS * DIM_HEAD
#define NB        8
#define NSEQ      8192
#define NLAT      64
#define NKV       (NSEQ + NLAT)   // 8256 (multiple of 32)
#define SCALE_ATTN 8.0f
#define LN_EPS    1e-5f
#define L2_EPS    1e-12f
#define MAXNEG    (-3.4028234663852886e+38f)

typedef __bf16 bf16;
typedef __attribute__((ext_vector_type(16))) __bf16 v16bf;
typedef __attribute__((ext_vector_type(8)))  __bf16 v8bf;
typedef __attribute__((ext_vector_type(8)))  float  v8f;
typedef __attribute__((ext_vector_type(4)))  uint32_t su32x4;
typedef __attribute__((ext_vector_type(8)))  uint32_t su32x8;

// ---------------- scalar conversions (bit-exact RNE) ----------------
__device__ __forceinline__ bf16 f2bf(float f) {
  union { float f; uint32_t u; } a; a.f = f;
  uint32_t r = a.u + 0x7FFFu + ((a.u >> 16) & 1u);
  union { uint16_t u; bf16 b; } o; o.u = (uint16_t)(r >> 16);
  return o.b;
}

// ---------------- WMMA helpers ----------------
// A/B fragment (16x32 bf16) from a row-major [ld] bf16 matrix.
// lane&15 -> row (M for A, N for B-as-transposed); element e ->
//   K = k0 + (e>>3)*16 + (lane>>4)*8 + (e&7)   (ISA 7.12.2 16-bit A layout)
__device__ __forceinline__ v16bf load_frag_bf(const bf16* p, int ld, int r0, int k0) {
  const int lane = threadIdx.x & 31;
  const bf16* q = p + (size_t)(r0 + (lane & 15)) * ld + (size_t)(k0 + ((lane >> 4) << 3));
  union { v16bf f; struct { v8bf lo, hi; } h; } u;
  u.h.lo = *(const v8bf*)(q);
  u.h.hi = *(const v8bf*)(q + 16);
  return u.f;
}

// Same fragment but sourced from f32 memory, converting to bf16 on the fly.
__device__ __forceinline__ v16bf load_frag_f32(const float* p, int ld, int r0, int k0) {
  const int lane = threadIdx.x & 31;
  const float* q = p + (size_t)(r0 + (lane & 15)) * ld + (size_t)(k0 + ((lane >> 4) << 3));
  v16bf f;
#pragma unroll
  for (int i = 0; i < 8; ++i) { f[i] = f2bf(q[i]); f[8 + i] = f2bf(q[16 + i]); }
  return f;
}

__device__ __forceinline__ v8f wmma_bf16(v16bf a, v16bf b, v8f c) {
  return __builtin_amdgcn_wmma_f32_16x16x32_bf16(false, a, false, b, (short)0, c, false, false);
}

// ---------------- TDM: DMA a [rows x cols] bf16 tile (row stride ld elems)
// from global memory into LDS. 2-D descriptor, groups 2/3 disabled (NULL).
// Issue from ONE wave only; caller must s_wait_tensorcnt + barrier.
__device__ __forceinline__ void tdm_load_tile_bf16(const bf16* gsrc, uint32_t lds_byte_off,
                                                   uint32_t cols, uint32_t rows, uint32_t ld) {
  const uint64_t gaddr = (uint64_t)(uintptr_t)gsrc;
  su32x4 g0;
  g0[0] = 1u;                                            // count=1 (valid), user mode
  g0[1] = lds_byte_off;                                  // lds_addr
  g0[2] = (uint32_t)gaddr;                               // global_addr[31:0]
  g0[3] = (uint32_t)((gaddr >> 32) & 0x1FFFFFFu) | (2u << 30);  // addr[56:32] | type=2
  su32x8 g1;
  g1[0] = (1u << 16);                                    // wg_mask=0, data_size=1 (2B)
  g1[1] = (cols & 0xFFFFu) << 16;                        // tensor_dim0[15:0]
  g1[2] = ((cols >> 16) & 0xFFFFu) | ((rows & 0xFFFFu) << 16);   // dim0 hi | tensor_dim1 lo
  g1[3] = ((rows >> 16) & 0xFFFFu) | ((cols & 0xFFFFu) << 16);   // dim1 hi | tile_dim0
  g1[4] = rows & 0xFFFFu;                                // tile_dim1 | tile_dim2=0
  g1[5] = ld;                                            // tensor_dim0_stride[31:0]
  g1[6] = 0u;                                            // stride hi | dim1_stride lo
  g1[7] = 0u;                                            // dim1_stride hi
  asm volatile("tensor_load_to_lds %0, %1" :: "s"(g0), "s"(g1) : "memory");
}

// ---------------- LayerNorm (row length = DIM), out bf16 or f32 ----------------
__global__ void ln_kernel(const float* __restrict__ x, const float* __restrict__ w,
                          const float* __restrict__ bias, bf16* __restrict__ obf,
                          float* __restrict__ of32) {
  const int row = blockIdx.x;
  const float* xr = x + (size_t)row * DIM;
  float v[4];
  float s = 0.f;
#pragma unroll
  for (int i = 0; i < 4; ++i) { v[i] = xr[threadIdx.x + i * 256]; s += v[i]; }
  __shared__ float red[8];
  for (int o = 16; o > 0; o >>= 1) s += __shfl_xor(s, o, 32);
  if ((threadIdx.x & 31) == 0) red[threadIdx.x >> 5] = s;
  __syncthreads();
  float mean = 0.f;
#pragma unroll
  for (int i = 0; i < 8; ++i) mean += red[i];
  mean *= (1.0f / DIM);
  __syncthreads();
  float s2 = 0.f;
#pragma unroll
  for (int i = 0; i < 4; ++i) { float d = v[i] - mean; s2 += d * d; }
  for (int o = 16; o > 0; o >>= 1) s2 += __shfl_xor(s2, o, 32);
  if ((threadIdx.x & 31) == 0) red[threadIdx.x >> 5] = s2;
  __syncthreads();
  float var = 0.f;
#pragma unroll
  for (int i = 0; i < 8; ++i) var += red[i];
  var *= (1.0f / DIM);
  const float rs = rsqrtf(var + LN_EPS);
#pragma unroll
  for (int i = 0; i < 4; ++i) {
    const int c = threadIdx.x + i * 256;
    const float o = (v[i] - mean) * rs * w[c] + bias[c];
    if (obf) obf[(size_t)row * DIM + c] = f2bf(o);
    else     of32[(size_t)row * DIM + c] = o;
  }
}

// ---------------- weight transpose + bf16 convert: src f32 [R][C] -> dst bf16 [C][R] ----
__global__ void transpose_to_bf16(const float* __restrict__ src, bf16* __restrict__ dst,
                                  int R, int C) {
  const int idx = blockIdx.x * blockDim.x + threadIdx.x;
  if (idx >= R * C) return;
  const int c = idx / R, r = idx - c * R;
  dst[idx] = f2bf(src[(size_t)r * C + c]);
}

// ---------------- q = LN(latents) @ Wq, one wave per 16x16 tile -------------
__global__ void qgemm_kernel(const bf16* __restrict__ latln, const bf16* __restrict__ WqT,
                             float* __restrict__ qf32) {
  const int idx = blockIdx.x;               // NB * 4 * 32
  const int b = idx >> 7, rem = idx & 127;
  const int rt = rem >> 5, ct = rem & 31;
  const bf16* A = latln + ((size_t)b * NLAT + rt * 16) * DIM;
  v8f acc = {};
  for (int k = 0; k < DIM; k += 32) {
    v16bf a  = load_frag_bf(A, DIM, 0, k);
    v16bf bb = load_frag_bf(WqT, DIM, ct * 16, k);
    acc = wmma_bf16(a, bb, acc);
  }
  const int lane = threadIdx.x & 31;
  const int half = lane >> 4, n = lane & 15;
#pragma unroll
  for (int j = 0; j < 8; ++j) {
    const int row = rt * 16 + half * 8 + j;
    qf32[((size_t)b * NLAT + row) * INNER + ct * 16 + n] = acc[j];
  }
}

// ---------------- per-head L2 norm of q, write bf16 [b,h,64,64] -------------
__global__ void qnorm_kernel(const float* __restrict__ qf32, const float* __restrict__ q_scale,
                             bf16* __restrict__ qbf) {
  const int idx = blockIdx.x;               // NB*NLAT*HEADS
  const int b = idx >> 9, pos = (idx >> 3) & 63, h = idx & 7;
  const int d = threadIdx.x;                // 64 threads
  const float v = qf32[((size_t)b * NLAT + pos) * INNER + h * DIM_HEAD + d];
  float s = v * v;
  for (int o = 16; o > 0; o >>= 1) s += __shfl_xor(s, o, 32);
  __shared__ float part[2];
  if ((threadIdx.x & 31) == 0) part[threadIdx.x >> 5] = s;
  __syncthreads();
  const float tot = part[0] + part[1];
  const float inv = 1.0f / fmaxf(sqrtf(tot), L2_EPS);
  qbf[(((size_t)(b * HEADS + h)) * NLAT + pos) * DIM_HEAD + d] = f2bf(v * inv * q_scale[d]);
}

// ---------------- fused kv GEMM (the 138-GFLOP one) -------------------------
// grid = NB * (NKV/16) * 16 bands, block = 128 (4 waves).
// The 16x1024 bf16 A tile (32 KB) is DMA'd into LDS once per block by the
// Tensor Data Mover; all 4 waves then read A-fragments via ds_load_b128
// (removes the 4x redundant global A traffic). B streams from L2.
// band<8  -> k head `band`  : l2norm over 64 dims (cross-wave LDS reduce) * k_scale
// band>=8 -> v head `band-8`: stored transposed [b,h][d][key] for attn B-fragments
__global__ void kvgemm_kernel(const bf16* __restrict__ xln, const bf16* __restrict__ latln,
                              const bf16* __restrict__ WkvT, const float* __restrict__ k_scale,
                              bf16* __restrict__ kbf, bf16* __restrict__ vtbf) {
  const int ROWTILES = NKV / 16;            // 516
  const int idx = blockIdx.x;
  const int b    = idx / (ROWTILES * 16);
  const int rem  = idx % (ROWTILES * 16);
  const int rt   = rem >> 4;
  const int band = rem & 15;
  const int row0 = rt * 16;
  const int w    = threadIdx.x >> 5;
  const int lane = threadIdx.x & 31;
  const int col0 = band * 64 + w * 16;      // column into the 1024-wide kv output
  const bf16* A = (row0 < NSEQ)
      ? xln   + ((size_t)b * NSEQ + row0) * DIM
      : latln + ((size_t)b * NLAT + (row0 - NSEQ)) * DIM;

  __shared__ bf16 Atile[16 * DIM];          // 32 KB staging tile
  if (threadIdx.x < 32) {                   // one wave issues the TDM transfer
    tdm_load_tile_bf16(A, (uint32_t)(uintptr_t)&Atile[0], DIM, 16, DIM);
    __builtin_amdgcn_s_wait_tensorcnt(0);
  }
  __syncthreads();

  const bf16* Al = &Atile[0];
  v8f acc = {};
  for (int k = 0; k < DIM; k += 32) {
    v16bf a  = load_frag_bf(Al, DIM, 0, k);          // ds_load_b128 x2
    v16bf bb = load_frag_bf(WkvT, DIM, col0, k);     // global_load_b128 x2
    acc = wmma_bf16(a, bb, acc);
  }
  const int half = lane >> 4, n = lane & 15;
  const int h = band & 7;
  const size_t bh = (size_t)(b * HEADS + h);
  if (band < 8) {
    __shared__ float norm2[4][16];
#pragma unroll
    for (int j = 0; j < 8; ++j) {
      float s = acc[j] * acc[j];
      for (int o = 8; o > 0; o >>= 1) s += __shfl_xor(s, o, 16);
      if (n == 0) norm2[w][half * 8 + j] = s;
    }
    __syncthreads();
#pragma unroll
    for (int j = 0; j < 8; ++j) {
      const int M = half * 8 + j;
      const float tot = norm2[0][M] + norm2[1][M] + norm2[2][M] + norm2[3][M];
      const float inv = 1.0f / fmaxf(sqrtf(tot), L2_EPS);
      const int d = w * 16 + n;
      kbf[(bh * NKV + row0 + M) * DIM_HEAD + d] = f2bf(acc[j] * inv * k_scale[d]);
    }
  } else {
#pragma unroll
    for (int j = 0; j < 8; ++j) {
      const int M = half * 8 + j;
      const int d = w * 16 + n;
      vtbf[(bh * DIM_HEAD + d) * NKV + row0 + M] = f2bf(acc[j]);
    }
  }
}

// ---------------- flash attention per (b,h): 4 waves x 16 query rows --------
__global__ void attn_kernel(const bf16* __restrict__ qbf, const bf16* __restrict__ kbf,
                            const bf16* __restrict__ vtbf, const int* __restrict__ mask,
                            float* __restrict__ oh) {
  const int b = blockIdx.x >> 3, h = blockIdx.x & 7;
  const int w = threadIdx.x >> 5;           // wave -> query-row tile
  const int lane = threadIdx.x & 31;
  const int half = lane >> 4, n = lane & 15;
  const size_t bh = (size_t)(b * HEADS + h);
  const bf16* qbase = qbf  + bh * NLAT * DIM_HEAD;
  const bf16* kbase = kbf  + bh * (size_t)NKV * DIM_HEAD;
  const bf16* vtb   = vtbf + bh * (size_t)DIM_HEAD * NKV;
  const v16bf q0 = load_frag_bf(qbase, DIM_HEAD, w * 16, 0);
  const v16bf q1 = load_frag_bf(qbase, DIM_HEAD, w * 16, 32);
  v8f acc0 = {}, acc1 = {}, acc2 = {}, acc3 = {};
  float mrow[8], lrow[8];
#pragma unroll
  for (int j = 0; j < 8; ++j) { mrow[j] = MAXNEG; lrow[j] = 0.f; }
  __shared__ bf16 pshare[4][16][32];        // per-wave P tile, C-layout -> A-layout relay

  for (int key0 = 0; key0 < NKV; key0 += 32) {
    v8f s0 = {}, s1 = {};
    s0 = wmma_bf16(q0, load_frag_bf(kbase, DIM_HEAD, key0, 0),       s0);
    s0 = wmma_bf16(q1, load_frag_bf(kbase, DIM_HEAD, key0, 32),      s0);
    s1 = wmma_bf16(q0, load_frag_bf(kbase, DIM_HEAD, key0 + 16, 0),  s1);
    s1 = wmma_bf16(q1, load_frag_bf(kbase, DIM_HEAD, key0 + 16, 32), s1);

    const int keyA = key0 + n, keyB = key0 + 16 + n;
    const bool mA = (keyA >= NSEQ) || (mask[(size_t)b * NSEQ + keyA] != 0);
    const bool mB = (keyB >= NSEQ) || (mask[(size_t)b * NSEQ + keyB] != 0);
#pragma unroll
    for (int j = 0; j < 8; ++j) {
      s0[j] = mA ? s0[j] * SCALE_ATTN : MAXNEG;
      s1[j] = mB ? s1[j] * SCALE_ATTN : MAXNEG;
    }
    // online softmax (per row = half*8+j, reduced over 16-lane half-groups)
#pragma unroll
    for (int j = 0; j < 8; ++j) {
      float vm = fmaxf(s0[j], s1[j]);
      for (int o = 8; o > 0; o >>= 1) vm = fmaxf(vm, __shfl_xor(vm, o, 16));
      const float nm   = fmaxf(mrow[j], vm);
      const float corr = __expf(mrow[j] - nm);
      const float p0 = __expf(s0[j] - nm);
      const float p1 = __expf(s1[j] - nm);
      float ps = p0 + p1;
      for (int o = 8; o > 0; o >>= 1) ps += __shfl_xor(ps, o, 16);
      lrow[j] = lrow[j] * corr + ps;
      mrow[j] = nm;
      acc0[j] *= corr; acc1[j] *= corr; acc2[j] *= corr; acc3[j] *= corr;
      const int M = half * 8 + j;
      pshare[w][M][n]      = f2bf(p0);
      pshare[w][M][16 + n] = f2bf(p1);
    }
    const v16bf pf = load_frag_bf(&pshare[w][0][0], 32, 0, 0);
    acc0 = wmma_bf16(pf, load_frag_bf(vtb, NKV, 0,  key0), acc0);
    acc1 = wmma_bf16(pf, load_frag_bf(vtb, NKV, 16, key0), acc1);
    acc2 = wmma_bf16(pf, load_frag_bf(vtb, NKV, 32, key0), acc2);
    acc3 = wmma_bf16(pf, load_frag_bf(vtb, NKV, 48, key0), acc3);
  }
#pragma unroll
  for (int j = 0; j < 8; ++j) {
    const int M = half * 8 + j;
    const float inv = 1.0f / lrow[j];
    const size_t orow = ((size_t)b * NLAT + w * 16 + M) * INNER + h * DIM_HEAD;
    oh[orow + 0  + n] = acc0[j] * inv;
    oh[orow + 16 + n] = acc1[j] * inv;
    oh[orow + 32 + n] = acc2[j] * inv;
    oh[orow + 48 + n] = acc3[j] * inv;
  }
}

// ---------------- out @ Wo (pre-LN) -----------------------------------------
__global__ void ogemm_kernel(const float* __restrict__ oh, const bf16* __restrict__ WoT,
                             float* __restrict__ tmp) {
  const int rt = blockIdx.x >> 4;           // 32 row tiles (NB*NLAT/16)
  const int cg = blockIdx.x & 15;           // 16 col groups of 4 waves
  const int w  = threadIdx.x >> 5;
  const int ct = cg * 4 + w;
  const float* A = oh + (size_t)rt * 16 * INNER;
  v8f acc = {};
  for (int k = 0; k < INNER; k += 32) {
    v16bf a  = load_frag_f32(A, INNER, 0, k);
    v16bf bb = load_frag_bf(WoT, INNER, ct * 16, k);
    acc = wmma_bf16(a, bb, acc);
  }
  const int lane = threadIdx.x & 31, half = lane >> 4, n = lane & 15;
#pragma unroll
  for (int j = 0; j < 8; ++j)
    tmp[((size_t)rt * 16 + half * 8 + j) * DIM + ct * 16 + n] = acc[j];
}

// ---------------- launcher ---------------------------------------------------
extern "C" void kernel_launch(void* const* d_in, const int* in_sizes, int n_in,
                              void* d_out, int out_size, void* d_ws, size_t ws_size,
                              hipStream_t stream) {
  (void)in_sizes; (void)n_in; (void)out_size; (void)ws_size;
  const float* x       = (const float*)d_in[0];
  const float* lat     = (const float*)d_in[1];
  const int*   mask    = (const int*)d_in[2];
  const float* ln_x_w  = (const float*)d_in[3];
  const float* ln_x_b  = (const float*)d_in[4];
  const float* ln_l_w  = (const float*)d_in[5];
  const float* ln_l_b  = (const float*)d_in[6];
  const float* Wq      = (const float*)d_in[7];
  const float* Wkv     = (const float*)d_in[8];
  const float* q_scale = (const float*)d_in[9];
  const float* k_scale = (const float*)d_in[10];
  const float* Wo      = (const float*)d_in[11];
  const float* ln_o_w  = (const float*)d_in[12];
  const float* ln_o_b  = (const float*)d_in[13];

  char* ws = (char*)d_ws;
  size_t off = 0;
  auto wsalloc = [&](size_t bytes) -> void* {
    void* p = ws + off;
    off = (off + bytes + 255) & ~(size_t)255;
    return p;
  };
  bf16*  XLN   = (bf16*) wsalloc((size_t)NB * NSEQ * DIM * 2);
  bf16*  LATLN = (bf16*) wsalloc((size_t)NB * NLAT * DIM * 2);
  bf16*  WQT   = (bf16*) wsalloc((size_t)INNER * DIM * 2);
  bf16*  WKVT  = (bf16*) wsalloc((size_t)DIM * DIM * 2);
  bf16*  WOT   = (bf16*) wsalloc((size_t)DIM * INNER * 2);
  float* QF32  = (float*)wsalloc((size_t)NB * NLAT * INNER * 4);
  bf16*  QBF   = (bf16*) wsalloc((size_t)NB * HEADS * NLAT * DIM_HEAD * 2);
  bf16*  KBF   = (bf16*) wsalloc((size_t)NB * HEADS * NKV * DIM_HEAD * 2);
  bf16*  VTBF  = (bf16*) wsalloc((size_t)NB * HEADS * DIM_HEAD * NKV * 2);
  float* OH    = (float*)wsalloc((size_t)NB * NLAT * INNER * 4);
  float* TMP   = (float*)wsalloc((size_t)NB * NLAT * DIM * 4);

  ln_kernel<<<NB * NSEQ, 256, 0, stream>>>(x,   ln_x_w, ln_x_b, XLN,   nullptr);
  ln_kernel<<<NB * NLAT, 256, 0, stream>>>(lat, ln_l_w, ln_l_b, LATLN, nullptr);
  transpose_to_bf16<<<(DIM * INNER + 255) / 256, 256, 0, stream>>>(Wq,  WQT,  DIM,  INNER);
  transpose_to_bf16<<<(DIM * DIM   + 255) / 256, 256, 0, stream>>>(Wkv, WKVT, DIM,  DIM);
  transpose_to_bf16<<<(INNER * DIM + 255) / 256, 256, 0, stream>>>(Wo,  WOT,  INNER, DIM);
  qgemm_kernel<<<NB * 4 * 32, 32, 0, stream>>>(LATLN, WQT, QF32);
  qnorm_kernel<<<NB * NLAT * HEADS, 64, 0, stream>>>(QF32, q_scale, QBF);
  kvgemm_kernel<<<NB * (NKV / 16) * 16, 128, 0, stream>>>(XLN, LATLN, WKVT, k_scale, KBF, VTBF);
  attn_kernel<<<NB * HEADS, 128, 0, stream>>>(QBF, KBF, VTBF, mask, OH);
  ogemm_kernel<<<32 * 16, 128, 0, stream>>>(OH, WOT, TMP);
  ln_kernel<<<NB * NLAT, 256, 0, stream>>>(TMP, ln_o_w, ln_o_b, nullptr, (float*)d_out);
}
